// BasicAttentionModel_7430293422605
// MI455X (gfx1250) — matrix operand order repaired and checked by reading, not compiled, not run on hardware
//
#include <hip/hip_runtime.h>
#include <hip/hip_bf16.h>

// ---------------------------------------------------------------------------
// GAT (3 layers, 4 heads, head-avg) + edge-predictor MLP for gfx1250 (MI455X).
// MLP (7.9 GMAC, compute-dominant) runs on v_wmma_f32_16x16x32_f16 with
// LDS-staged fragments (ds_load_b128) and vectorized L2-resident gathers.
// Input order assumed = setup_inputs() dict insertion order, recursively:
//  0 x[50000*16] 1 e[800000*10] 2 edge_index[2*800000](int)
//  3..6  bn_node g,b,mean,var   7..10 bn_edge g,b,mean,var
//  11..14 gat1 W,att_src,att_dst,bias   15..18 gat2   19..22 gat3
//  23..28 mlp l1.W,l1.b,l2.W,l2.b,l3.W,l3.b
// Workspace need ~132 MB.
// ---------------------------------------------------------------------------

typedef __attribute__((ext_vector_type(16))) _Float16 v16h;
typedef __attribute__((ext_vector_type(8)))  float    v8f;

union Frag16 { v16h v; uint4 q[2]; _Float16 h[16]; };

static __device__ __forceinline__ v8f vzero8() {
  v8f z;
#pragma unroll
  for (int i = 0; i < 8; ++i) z[i] = 0.f;
  return z;
}

static __device__ __forceinline__ float leakyf(float v, float s) {
  return v > 0.f ? v : s * v;
}

// order-preserving float<->uint mapping for atomicMax on floats
static __device__ __forceinline__ unsigned fpun(float f) {
  unsigned u = __float_as_uint(f);
  return (u & 0x80000000u) ? ~u : (u | 0x80000000u);
}
static __device__ __forceinline__ float funpun(unsigned u) {
  return (u & 0x80000000u) ? __uint_as_float(u & 0x7FFFFFFFu)
                           : __uint_as_float(~u);
}

static __device__ __forceinline__ void edge_sd(const int* ei, int idx, int E,
                                               int& s, int& d) {
  if (idx < E) { s = ei[idx]; d = ei[E + idx]; }
  else         { s = d = idx - E; }          // self-loop
}

// pack 8 f32 -> 8 f16 in one uint4 (for ds_store_b128)
static __device__ __forceinline__ uint4 pack8(const float4& a, const float4& b) {
  union { _Float16 h[8]; uint4 q; } u;
  u.h[0] = (_Float16)a.x; u.h[1] = (_Float16)a.y;
  u.h[2] = (_Float16)a.z; u.h[3] = (_Float16)a.w;
  u.h[4] = (_Float16)b.x; u.h[5] = (_Float16)b.y;
  u.h[6] = (_Float16)b.z; u.h[7] = (_Float16)b.w;
  return u.q;
}

// ---------------------------------------------------------------------------
// prep: fold bn_edge into W1 -> W1t (f16, [64 cols][160 K], K-padded) + bias1'
// ---------------------------------------------------------------------------
__global__ void prep_kernel(const float* __restrict__ W1, const float* __restrict__ b1,
                            const float* __restrict__ eg, const float* __restrict__ eb,
                            const float* __restrict__ em, const float* __restrict__ ev,
                            _Float16* __restrict__ W1t, float* __restrict__ bias1) {
  for (int idx = threadIdx.x; idx < 64 * 160; idx += 256) {
    int n = idx / 160, k = idx - n * 160;
    float v = 0.f;
    if (k < 128) v = W1[k * 64 + n];
    else if (k < 138) {
      int j = k - 128;
      float s = eg[j] * rsqrtf(ev[j] + 1e-5f);
      v = W1[k * 64 + n] * s;
    }
    W1t[idx] = (_Float16)v;   // idx == n*160 + k
  }
  if (threadIdx.x < 64) {
    int n = threadIdx.x;
    float s = b1[n];
    for (int j = 0; j < 10; ++j) {
      float sj = eg[j] * rsqrtf(ev[j] + 1e-5f);
      s += (eb[j] - em[j] * sj) * W1[(128 + j) * 64 + n];
    }
    bias1[n] = s;
  }
}

// ---------------------------------------------------------------------------
// BatchNorm (inference) on node features, 16 channels
// ---------------------------------------------------------------------------
__global__ void bn_node_kernel(const float* __restrict__ x,
                               const float* __restrict__ g, const float* __restrict__ b,
                               const float* __restrict__ mean, const float* __restrict__ var,
                               float* __restrict__ xout, int N) {
  int i = blockIdx.x * 256 + threadIdx.x;
  if (i >= N * 16) return;
  int c = i & 15;
  float s = g[c] * rsqrtf(var[c] + 1e-5f);
  xout[i] = (x[i] - mean[c]) * s + b[c];
}

// ---------------------------------------------------------------------------
// GAT per-node transform: h = x @ W, att logits; init m/den/accum
// ---------------------------------------------------------------------------
template <int CIN, int COUT>
__global__ __launch_bounds__(256) void gat_transform(
    const float* __restrict__ xin, const float* __restrict__ W,
    const float* __restrict__ att_src, const float* __restrict__ att_dst,
    float* __restrict__ hbuf, float* __restrict__ a_src, float* __restrict__ a_dst,
    float* __restrict__ accum, unsigned* __restrict__ mbuf, float* __restrict__ den,
    int N) {
  __shared__ float sW[CIN * 4 * COUT];
  for (int i = threadIdx.x; i < CIN * 4 * COUT; i += 256) sW[i] = W[i];
  __syncthreads();
  int tid = blockIdx.x * 256 + threadIdx.x;
  int nd = tid >> 2, head = tid & 3;
  if (nd >= N) return;
  float xr[CIN];
#pragma unroll
  for (int k = 0; k < CIN; ++k) xr[k] = xin[(size_t)nd * CIN + k];
  float acc[COUT];
#pragma unroll
  for (int c = 0; c < COUT; ++c) acc[c] = 0.f;
  for (int k = 0; k < CIN; ++k) {
    float xk = xr[k];
    const float* wrow = &sW[k * 4 * COUT + head * COUT];
#pragma unroll
    for (int c = 0; c < COUT; ++c) acc[c] = fmaf(xk, wrow[c], acc[c]);
  }
  float as = 0.f, ad = 0.f;
#pragma unroll
  for (int c = 0; c < COUT; ++c) {
    as = fmaf(acc[c], att_src[head * COUT + c], as);
    ad = fmaf(acc[c], att_dst[head * COUT + c], ad);
  }
  float* hrow = &hbuf[(size_t)nd * 4 * COUT + head * COUT];
  float* arow = &accum[(size_t)nd * 4 * COUT + head * COUT];
#pragma unroll
  for (int c = 0; c < COUT; ++c) { hrow[c] = acc[c]; arow[c] = 0.f; }
  a_src[nd * 4 + head] = as;
  a_dst[nd * 4 + head] = ad;
  mbuf[nd * 4 + head]  = 0u;     // < fpun of any real value; self-loop guarantees update
  den[nd * 4 + head]   = 0.f;
}

// ---------------------------------------------------------------------------
// segment max of leaky logits over dst
// ---------------------------------------------------------------------------
__global__ void gat_edge_max(const int* __restrict__ ei,
                             const float* __restrict__ a_src, const float* __restrict__ a_dst,
                             unsigned* __restrict__ mbuf, int E, int N) {
  int idx = blockIdx.x * 256 + threadIdx.x;
  if (idx >= E + N) return;
  int s, d; edge_sd(ei, idx, E, s, d);
  float4 as = *(const float4*)(a_src + (size_t)s * 4);
  float4 ad = *(const float4*)(a_dst + (size_t)d * 4);
  atomicMax(&mbuf[d * 4 + 0], fpun(leakyf(as.x + ad.x, 0.2f)));
  atomicMax(&mbuf[d * 4 + 1], fpun(leakyf(as.y + ad.y, 0.2f)));
  atomicMax(&mbuf[d * 4 + 2], fpun(leakyf(as.z + ad.z, 0.2f)));
  atomicMax(&mbuf[d * 4 + 3], fpun(leakyf(as.w + ad.w, 0.2f)));
}

// ---------------------------------------------------------------------------
// segment sum of exp(logit - m)
// ---------------------------------------------------------------------------
__global__ void gat_edge_sum(const int* __restrict__ ei,
                             const float* __restrict__ a_src, const float* __restrict__ a_dst,
                             const unsigned* __restrict__ mbuf, float* __restrict__ den,
                             int E, int N) {
  int idx = blockIdx.x * 256 + threadIdx.x;
  if (idx >= E + N) return;
  int s, d; edge_sd(ei, idx, E, s, d);
  float4 as = *(const float4*)(a_src + (size_t)s * 4);
  float4 ad = *(const float4*)(a_dst + (size_t)d * 4);
  float l[4] = {leakyf(as.x + ad.x, 0.2f), leakyf(as.y + ad.y, 0.2f),
                leakyf(as.z + ad.z, 0.2f), leakyf(as.w + ad.w, 0.2f)};
#pragma unroll
  for (int h = 0; h < 4; ++h) {
    float m = funpun(mbuf[d * 4 + h]);
    atomicAdd(&den[d * 4 + h], expf(l[h] - m));
  }
}

// ---------------------------------------------------------------------------
// weighted scatter: accum[dst] += h[src] * alpha ; 1 thread per (edge, head)
// ---------------------------------------------------------------------------
template <int COUT>
__global__ void gat_edge_scatter(const int* __restrict__ ei,
                                 const float* __restrict__ a_src, const float* __restrict__ a_dst,
                                 const unsigned* __restrict__ mbuf, const float* __restrict__ den,
                                 const float* __restrict__ hbuf, float* __restrict__ accum,
                                 int E, int N) {
  int tid = blockIdx.x * 256 + threadIdx.x;
  int idx = tid >> 2, head = tid & 3;
  if (idx >= E + N) return;
  int s, d; edge_sd(ei, idx, E, s, d);
  float l = leakyf(a_src[s * 4 + head] + a_dst[d * 4 + head], 0.2f);
  float m = funpun(mbuf[d * 4 + head]);
  float alpha = expf(l - m) / (den[d * 4 + head] + 1e-16f);
  const float4* hp = (const float4*)(hbuf + (size_t)s * 4 * COUT + head * COUT);
  float* ap = accum + (size_t)d * 4 * COUT + head * COUT;
#pragma unroll
  for (int c4 = 0; c4 < COUT / 4; ++c4) {
    float4 v = hp[c4];
    atomicAdd(ap + 4 * c4 + 0, v.x * alpha);
    atomicAdd(ap + 4 * c4 + 1, v.y * alpha);
    atomicAdd(ap + 4 * c4 + 2, v.z * alpha);
    atomicAdd(ap + 4 * c4 + 3, v.w * alpha);
  }
}

// ---------------------------------------------------------------------------
// head average + bias
// ---------------------------------------------------------------------------
template <int COUT>
__global__ void gat_finalize(const float* __restrict__ accum, const float* __restrict__ bias,
                             float* __restrict__ xout, int N) {
  int i = blockIdx.x * 256 + threadIdx.x;
  if (i >= N * COUT) return;
  int nd = i / COUT, c = i - nd * COUT;
  const float* a = accum + (size_t)nd * 4 * COUT + c;
  float s = a[0 * COUT] + a[1 * COUT] + a[2 * COUT] + a[3 * COUT];
  xout[i] = 0.25f * s + bias[c];
}

// ---------------------------------------------------------------------------
// Edge-predictor MLP on WMMA.
// 128 threads = 4 waves; each wave owns a 16-edge tile; block iterates
// MLP_ITER batches of 64 edges (weights staged once per block).
// Per tile: D[16x64]=A[16x160]@W1t (20 wmma), leaky; [16x64]@[64x16] (2 wmma),
// leaky; dot16 + sigmoid.
// ---------------------------------------------------------------------------
#define MLP_ITER 4

__global__ __launch_bounds__(128) void edge_mlp(
    const float* __restrict__ x3, const float* __restrict__ e,
    const int* __restrict__ ei,
    const _Float16* __restrict__ W1t, const float* __restrict__ bias1,
    const float* __restrict__ W2, const float* __restrict__ b2,
    const float* __restrict__ W3, const float* __restrict__ b3,
    float* __restrict__ out, int E) {
  __shared__ __align__(16) _Float16 sW1[64 * 160];   // B1, n-major
  __shared__ __align__(16) _Float16 sW2[16 * 64];    // B2, n-major
  __shared__ __align__(16) _Float16 sA[4][16 * 160]; // per-wave A tile
  __shared__ __align__(16) _Float16 sZ1[4][16 * 64];
  __shared__ float sZ2[4][16 * 16];
  __shared__ float sB1[64];
  __shared__ float sB2[16];
  __shared__ float sW3[16];

  const int t = threadIdx.x;

  // ---- stage weights once per block
  {
    const uint4* src = (const uint4*)W1t;
    uint4* dst = (uint4*)sW1;
    for (int i = t; i < (64 * 160 * 2) / 16; i += 128) dst[i] = src[i];
  }
  for (int i = t; i < 16 * 64; i += 128) {
    int n = i >> 6, k = i & 63;
    sW2[n * 64 + k] = (_Float16)W2[k * 16 + n];   // transpose + f16
  }
  if (t < 64) sB1[t] = bias1[t];
  if (t < 16) { sB2[t] = b2[t]; sW3[t] = W3[t]; }

  const int L = t & 31, w = t >> 5;
  const int m = L & 15, hi = (L >> 4) & 1;
  const int eb = blockIdx.x * (64 * MLP_ITER);
  const uint4 uz = uint4{0u, 0u, 0u, 0u};

  for (int it = 0; it < MLP_ITER; ++it) {
    // ---- gather 64 edges (2 threads/edge): [x[s](64) | x[t](64) | e(10) | pad]
    {
      int le = t >> 1, p = t & 1;
      int gw = le >> 4, gm = le & 15;
      int ed = eb + it * 64 + le;
      bool valid = ed < E;
      int s = 0, d2 = 0;
      if (valid) { s = ei[ed]; d2 = ei[E + ed]; }
      _Float16* row = &sA[gw][gm * 160];
      if (p == 0) {
        const float4* xs4 = (const float4*)(x3 + (size_t)s * 64);
#pragma unroll
        for (int i = 0; i < 8; ++i)
          *(uint4*)&row[8 * i] = valid ? pack8(xs4[2 * i], xs4[2 * i + 1]) : uz;
      } else {
        const float4* xt4 = (const float4*)(x3 + (size_t)d2 * 64);
#pragma unroll
        for (int i = 0; i < 8; ++i)
          *(uint4*)&row[64 + 8 * i] = valid ? pack8(xt4[2 * i], xt4[2 * i + 1]) : uz;
        float ef[10];
        if (valid) {
          const float2* ev2 = (const float2*)(e + (size_t)ed * 10);
#pragma unroll
          for (int i = 0; i < 5; ++i) { float2 v = ev2[i]; ef[2*i] = v.x; ef[2*i+1] = v.y; }
        } else {
#pragma unroll
          for (int i = 0; i < 10; ++i) ef[i] = 0.f;
        }
        union { _Float16 h[8]; uint4 q; } u1;
#pragma unroll
        for (int i = 0; i < 8; ++i) u1.h[i] = (_Float16)ef[i];
        *(uint4*)&row[128] = u1.q;
        union { _Float16 h[8]; uint4 q; } u2;
        u2.h[0] = (_Float16)ef[8]; u2.h[1] = (_Float16)ef[9];
#pragma unroll
        for (int i = 2; i < 8; ++i) u2.h[i] = (_Float16)0.f;
        *(uint4*)&row[136] = u2.q;
        *(uint4*)&row[144] = uz;
        *(uint4*)&row[152] = uz;
      }
    }
    __syncthreads();

    // ---- layer 1: 4 col-tiles x 5 K-chunks of v_wmma_f32_16x16x32_f16
    v8f acc[4];
#pragma unroll
    for (int nt = 0; nt < 4; ++nt) acc[nt] = vzero8();
#pragma unroll
    for (int c = 0; c < 5; ++c) {
      Frag16 a;  // A 16x32 f16 layout: 16 contiguous K at 32c+16*hi
      a.q[0] = *(const uint4*)&sA[w][m * 160 + 32 * c + 16 * hi];
      a.q[1] = *(const uint4*)&sA[w][m * 160 + 32 * c + 16 * hi + 8];
#pragma unroll
      for (int nt = 0; nt < 4; ++nt) {
        Frag16 b;  // B 32x16: lane col = m, 16 contiguous K at 32c+16*hi
        const uint4* bp = (const uint4*)&sW1[(nt * 16 + m) * 160 + 32 * c + 16 * hi];
        b.q[0] = bp[0]; b.q[1] = bp[1];
        acc[nt] = __builtin_amdgcn_wmma_f32_16x16x32_f16(
            false, a.v, false, b.v, (short)0, acc[nt], false, false);
      }
    }
#pragma unroll
    for (int nt = 0; nt < 4; ++nt) {
      int col = nt * 16 + m;
      float bc = sB1[col];
#pragma unroll
      for (int r = 0; r < 8; ++r) {
        float v = acc[nt][r] + bc;           // C layout: row = r + 8*hi, col = m
        v = v > 0.f ? v : 0.12f * v;
        sZ1[w][(r + 8 * hi) * 64 + col] = (_Float16)v;
      }
    }
    __syncthreads();

    // ---- layer 2: [16x64] @ [64x16]
    v8f acc2 = vzero8();
#pragma unroll
    for (int c = 0; c < 2; ++c) {
      Frag16 a;
      a.q[0] = *(const uint4*)&sZ1[w][m * 64 + 32 * c + 16 * hi];
      a.q[1] = *(const uint4*)&sZ1[w][m * 64 + 32 * c + 16 * hi + 8];
      Frag16 b;
      const uint4* bp = (const uint4*)&sW2[m * 64 + 32 * c + 16 * hi];
      b.q[0] = bp[0]; b.q[1] = bp[1];
      acc2 = __builtin_amdgcn_wmma_f32_16x16x32_f16(
          false, a.v, false, b.v, (short)0, acc2, false, false);
    }
    {
      float bc = sB2[m];
#pragma unroll
      for (int r = 0; r < 8; ++r) {
        float v = acc2[r] + bc;
        v = v > 0.f ? v : 0.12f * v;
        sZ2[w][(r + 8 * hi) * 16 + m] = v;
      }
    }
    __syncthreads();

    // ---- layer 3 + sigmoid
    if (L < 16) {
      int ed = eb + it * 64 + w * 16 + L;
      if (ed < E) {
        float s = b3[0];
#pragma unroll
        for (int n = 0; n < 16; ++n) s = fmaf(sZ2[w][L * 16 + n], sW3[n], s);
        out[ed] = 1.f / (1.f + expf(-s));
      }
    }
    __syncthreads();   // protect sA/sZ1/sZ2 for next iteration
  }
}

// ---------------------------------------------------------------------------
// host side
// ---------------------------------------------------------------------------
template <int CIN, int COUT>
static void run_gat(const float* xin, const float* W, const float* as_p, const float* ad_p,
                    const float* bias, float* xout, float* hbuf, float* a_src, float* a_dst,
                    float* accum, unsigned* mbuf, float* den, const int* ei,
                    int N, int E, hipStream_t stream) {
  gat_transform<CIN, COUT><<<(N * 4 + 255) / 256, 256, 0, stream>>>(
      xin, W, as_p, ad_p, hbuf, a_src, a_dst, accum, mbuf, den, N);
  int EN = E + N;
  gat_edge_max<<<(EN + 255) / 256, 256, 0, stream>>>(ei, a_src, a_dst, mbuf, E, N);
  gat_edge_sum<<<(EN + 255) / 256, 256, 0, stream>>>(ei, a_src, a_dst, mbuf, den, E, N);
  gat_edge_scatter<COUT><<<((size_t)EN * 4 + 255) / 256, 256, 0, stream>>>(
      ei, a_src, a_dst, mbuf, den, hbuf, accum, E, N);
  gat_finalize<COUT><<<((size_t)N * COUT + 255) / 256, 256, 0, stream>>>(accum, bias, xout, N);
}

extern "C" void kernel_launch(void* const* d_in, const int* in_sizes, int n_in,
                              void* d_out, int out_size, void* d_ws, size_t ws_size,
                              hipStream_t stream) {
  const float* x    = (const float*)d_in[0];
  const float* e    = (const float*)d_in[1];
  const int*   ei   = (const int*)d_in[2];
  const float* bn_g = (const float*)d_in[3];
  const float* bn_b = (const float*)d_in[4];
  const float* bn_m = (const float*)d_in[5];
  const float* bn_v = (const float*)d_in[6];
  const float* be_g = (const float*)d_in[7];
  const float* be_b = (const float*)d_in[8];
  const float* be_m = (const float*)d_in[9];
  const float* be_v = (const float*)d_in[10];
  const float* g1W  = (const float*)d_in[11];
  const float* g1as = (const float*)d_in[12];
  const float* g1ad = (const float*)d_in[13];
  const float* g1b  = (const float*)d_in[14];
  const float* g2W  = (const float*)d_in[15];
  const float* g2as = (const float*)d_in[16];
  const float* g2ad = (const float*)d_in[17];
  const float* g2b  = (const float*)d_in[18];
  const float* g3W  = (const float*)d_in[19];
  const float* g3as = (const float*)d_in[20];
  const float* g3ad = (const float*)d_in[21];
  const float* g3b  = (const float*)d_in[22];
  const float* W1   = (const float*)d_in[23];
  const float* b1   = (const float*)d_in[24];
  const float* W2   = (const float*)d_in[25];
  const float* b2   = (const float*)d_in[26];
  const float* W3   = (const float*)d_in[27];
  const float* b3   = (const float*)d_in[28];
  (void)n_in; (void)out_size; (void)ws_size;

  const int N = in_sizes[0] / 16;   // 50000
  const int E = in_sizes[1] / 10;   // 800000

  char* wsb = (char*)d_ws;
  size_t off = 0;
  auto take = [&](size_t bytes) -> char* {
    char* p = wsb + off;
    off = (off + bytes + 255) & ~(size_t)255;
    return p;
  };
  float*     bufX  = (float*)take((size_t)N * 64 * 4);
  float*     bufY  = (float*)take((size_t)N * 64 * 4);
  float*     hbuf  = (float*)take((size_t)N * 256 * 4);
  float*     accum = (float*)take((size_t)N * 256 * 4);
  float*     a_src = (float*)take((size_t)N * 4 * 4);
  float*     a_dst = (float*)take((size_t)N * 4 * 4);
  unsigned*  mbuf  = (unsigned*)take((size_t)N * 4 * 4);
  float*     den   = (float*)take((size_t)N * 4 * 4);
  _Float16*  W1t   = (_Float16*)take(64 * 160 * 2);
  float*     bias1 = (float*)take(64 * 4);

  prep_kernel<<<1, 256, 0, stream>>>(W1, b1, be_g, be_b, be_m, be_v, W1t, bias1);
  bn_node_kernel<<<(N * 16 + 255) / 256, 256, 0, stream>>>(x, bn_g, bn_b, bn_m, bn_v, bufX, N);

  run_gat<16, 16>(bufX, g1W, g1as, g1ad, g1b, bufY, hbuf, a_src, a_dst, accum, mbuf, den, ei, N, E, stream);
  run_gat<16, 32>(bufY, g2W, g2as, g2ad, g2b, bufX, hbuf, a_src, a_dst, accum, mbuf, den, ei, N, E, stream);
  run_gat<32, 64>(bufX, g3W, g3as, g3ad, g3b, bufY, hbuf, a_src, a_dst, accum, mbuf, den, ei, N, E, stream);

  const int edges_per_block = 64 * MLP_ITER;
  edge_mlp<<<(E + edges_per_block - 1) / edges_per_block, 128, 0, stream>>>(
      bufY, e, ei, W1t, bias1, W2, b2, W3, b3, (float*)d_out, E);
}
